// AdaptiveTimeDecayAttention_49039936586041
// MI455X (gfx1250) — matrix operand order repaired and checked by reading, not compiled, not run on hardware
//
#include <hip/hip_runtime.h>
#include <hip/hip_bf16.h>

typedef __attribute__((ext_vector_type(16))) __bf16 v16bf;
typedef __attribute__((ext_vector_type(8)))  float  v8f;

#define BB   2048
#define LL   200
#define DD   128
#define HH1  64
#define HH2  32
#define MT   13            // ceil(200/16) M-tiles

// ---- LDS carve (bytes) ----
#define OFF_W1   0         // 8 steps * 4 tiles * 32 lanes * 16 bf16 = 32768
#define OFF_W2   32768     // 2*2*32*16 bf16 = 4096
#define OFF_KEYS 36864     // 8*16*128 f32   = 65536
#define OFF_H    102400    // 8*16*64 bf16   = 16384
#define OFF_Q    118784    // 128 f32        = 512
#define OFF_QW1  119296    // 64 f32         = 256
#define OFF_SCR  119552    // 256 f32        = 1024
#define OFF_RED  120576    // 256 f32        = 1024
#define SMEM_BYTES 121600

__global__ void __launch_bounds__(256, 2)
atda_kernel(const float* __restrict__ query, const float* __restrict__ keys,
            const unsigned char* __restrict__ keys_mask,
            const float* __restrict__ W1, const float* __restrict__ b1,
            const float* __restrict__ a1p,
            const float* __restrict__ W2, const float* __restrict__ b2,
            const float* __restrict__ a2p,
            const float* __restrict__ W3, const float* __restrict__ b3p,
            const float* __restrict__ decayp,
            float* __restrict__ out)
{
  extern __shared__ __align__(256) char smem[];
  __bf16* w1p   = (__bf16*)(smem + OFF_W1);
  __bf16* w2p   = (__bf16*)(smem + OFF_W2);
  float*  keysb = (float*) (smem + OFF_KEYS);
  __bf16* hbuf  = (__bf16*)(smem + OFF_H);
  float*  qs    = (float*) (smem + OFF_Q);
  float*  qw1   = (float*) (smem + OFF_QW1);
  float*  scr   = (float*) (smem + OFF_SCR);
  float*  red   = (float*) (smem + OFF_RED);

  const int tid  = threadIdx.x;
  const int lane = tid & 31;
  const int wave = tid >> 5;
  const int b    = blockIdx.x;
  const int klo  = (lane < 16) ? 0 : 8;   // A-fragment K-half select
  const int nlo  = lane & 15;             // B/C fragment column within tile

  // ---- stage query row (needed by qw1 below) ----
  if (tid < 32)
    ((float4*)qs)[tid] = ((const float4*)(query + (size_t)b * DD))[tid];

  // ---- stage folded W1eff -> bf16, pre-swizzled to B-fragment layout ----
  // eff rows 0..127  : W1[k] + W1[384+k]   (keys and keys-q's k part)
  // eff rows 128..255: W1[256+k]           (keys*q part)
  for (int j = 0; j < 64; ++j) {
    int idx = tid + 256 * j;              // 0..16383 over [256][64]
    int k = idx >> 6, n = idx & 63;
    float w = (k < 128) ? (W1[k * 64 + n] + W1[(384 + k) * 64 + n])
                        : W1[(128 + k) * 64 + n];
    int s = k >> 5, kr = k & 31;
    int half = (kr >> 3) & 1, e = (kr & 7) | ((kr >> 4) << 3);
    int pl = half * 16 + (n & 15);
    int t = n >> 4;
    w1p[(((s * 4 + t) * 32 + pl) << 4) + e] = (__bf16)w;
  }
  // ---- stage W2 ([64][32]) ----
  for (int j = 0; j < 8; ++j) {
    int idx = tid + 256 * j;              // 0..2047
    int k = idx >> 5, n = idx & 31;
    int s = k >> 5, kr = k & 31;
    int half = (kr >> 3) & 1, e = (kr & 7) | ((kr >> 4) << 3);
    int pl = half * 16 + (n & 15);
    int t = n >> 4;
    w2p[(((s * 2 + t) * 32 + pl) << 4) + e] = (__bf16)W2[idx];
  }
  __syncthreads();

  // ---- q-dependent constant bias: qw1[n] = sum_d q[d]*(W1[128+d][n]-W1[384+d][n]) ----
  {
    int n = tid & 63, part = tid >> 6;    // 4 partials of 32 d's each
    float p = 0.f;
    for (int dd = part * 32; dd < part * 32 + 32; ++dd)
      p += qs[dd] * (W1[(128 + dd) * 64 + n] - W1[(384 + dd) * 64 + n]);
    red[tid] = p;
  }
  __syncthreads();
  if (tid < 64)
    qw1[tid] = red[tid] + red[64 + tid] + red[128 + tid] + red[192 + tid];
  __syncthreads();

  const float a1 = *a1p, a2 = *a2p, b3 = *b3p;
  const float w3a = W3[nlo];
  const float w3b = W3[16 + nlo];
  float b1v[4], b2v[2];
#pragma unroll
  for (int t = 0; t < 4; ++t) b1v[t] = b1[t * 16 + nlo] + qw1[t * 16 + nlo];
#pragma unroll
  for (int t = 0; t < 2; ++t) b2v[t] = b2[t * 16 + nlo];

  // =================== phase 1: MLP scores via WMMA ===================
  for (int iter = 0; iter < 2; ++iter) {
    const int tile = wave + 8 * iter;
    const bool active = tile < MT;

    if (active) {
      // stage 16 keys rows (f32) for this tile, zero-pad rows >= L
      float* kt = keysb + wave * (16 * DD);
      for (int j = 0; j < 16; ++j) {
        int idx = j * 32 + lane;          // 512 float4s
        int r = idx >> 5, c4 = idx & 31;
        int l = tile * 16 + r;
        float4 v = make_float4(0.f, 0.f, 0.f, 0.f);
        if (l < LL)
          v = ((const float4*)(keys + ((size_t)b * LL + l) * DD))[c4];
        ((float4*)(kt + r * DD))[c4] = v;
      }
    }
    __syncthreads();

    if (active) {
      const float* krow = keysb + wave * (16 * DD) + (lane & 15) * DD;

      // ---- layer 1 (folded): k @ W1eff_K  +  (k*q) @ W1eff_KQ ----
      v8f acc[4] = {};
#pragma unroll
      for (int c = 0; c < 4; ++c) {
        const int dbase = c * 32;
        const float4 kA0 = *(const float4*)(krow + dbase + klo);
        const float4 kA1 = *(const float4*)(krow + dbase + klo + 4);
        const float4 kB0 = *(const float4*)(krow + dbase + 16 + klo);
        const float4 kB1 = *(const float4*)(krow + dbase + 16 + klo + 4);
        const float4 qA0 = *(const float4*)(qs + dbase + klo);
        const float4 qA1 = *(const float4*)(qs + dbase + klo + 4);
        const float4 qB0 = *(const float4*)(qs + dbase + 16 + klo);
        const float4 qB1 = *(const float4*)(qs + dbase + 16 + klo + 4);
        const float kv[16] = {kA0.x,kA0.y,kA0.z,kA0.w, kA1.x,kA1.y,kA1.z,kA1.w,
                              kB0.x,kB0.y,kB0.z,kB0.w, kB1.x,kB1.y,kB1.z,kB1.w};
        const float qv[16] = {qA0.x,qA0.y,qA0.z,qA0.w, qA1.x,qA1.y,qA1.z,qA1.w,
                              qB0.x,qB0.y,qB0.z,qB0.w, qB1.x,qB1.y,qB1.z,qB1.w};
        v16bf afragK, afragKQ;
#pragma unroll
        for (int i = 0; i < 16; ++i) {
          afragK[i]  = (__bf16)kv[i];
          afragKQ[i] = (__bf16)(kv[i] * qv[i]);
        }
#pragma unroll
        for (int t = 0; t < 4; ++t) {
          const v16bf bfK =
              *(const v16bf*)(w1p + (((c * 4 + t) * 32 + lane) << 4));
          acc[t] = __builtin_amdgcn_wmma_f32_16x16x32_bf16(
              false, afragK, false, bfK, (short)0, acc[t], false, false);
        }
#pragma unroll
        for (int t = 0; t < 4; ++t) {
          const v16bf bfKQ =
              *(const v16bf*)(w1p + ((((4 + c) * 4 + t) * 32 + lane) << 4));
          acc[t] = __builtin_amdgcn_wmma_f32_16x16x32_bf16(
              false, afragKQ, false, bfKQ, (short)0, acc[t], false, false);
        }
      }

      // ---- PReLU + restage h[16x64] as bf16 for layer-2 A fragments ----
      __bf16* hb = hbuf + wave * (16 * HH1);
      const int mbase = (lane < 16) ? 0 : 8;
#pragma unroll
      for (int t = 0; t < 4; ++t) {
        int n = t * 16 + nlo;
#pragma unroll
        for (int i = 0; i < 8; ++i) {
          float h = acc[t][i] + b1v[t];
          h = (h > 0.f) ? h : a1 * h;
          hb[(mbase + i) * HH1 + n] = (__bf16)h;
        }
      }
      __builtin_amdgcn_wave_barrier();
      asm volatile("s_wait_dscnt 0" ::: "memory");   // wave-local LDS RAW

      // ---- layer 2: [16x64] x [64x32], 2 K-steps, 2 N-tiles ----
      v8f acc2[2] = {};
#pragma unroll
      for (int s = 0; s < 2; ++s) {
        const __bf16* hrow = hb + (lane & 15) * HH1 + s * 32;
        v16bf afrag;
#pragma unroll
        for (int i = 0; i < 8; ++i) {
          afrag[i]     = hrow[klo + i];
          afrag[8 + i] = hrow[klo + 16 + i];
        }
#pragma unroll
        for (int t = 0; t < 2; ++t) {
          const v16bf bfrag =
              *(const v16bf*)(w2p + (((s * 2 + t) * 32 + lane) << 4));
          acc2[t] = __builtin_amdgcn_wmma_f32_16x16x32_bf16(
              false, afrag, false, bfrag, (short)0, acc2[t], false, false);
        }
      }

      // ---- layer 3: PReLU, dot with W3, 16-lane shfl reduction ----
      float part[8];
#pragma unroll
      for (int i = 0; i < 8; ++i) {
        float h0 = acc2[0][i] + b2v[0]; h0 = (h0 > 0.f) ? h0 : a2 * h0;
        float h1 = acc2[1][i] + b2v[1]; h1 = (h1 > 0.f) ? h1 : a2 * h1;
        part[i] = h0 * w3a + h1 * w3b;
      }
#pragma unroll
      for (int i = 0; i < 8; ++i) {
        float r = part[i];
        r += __shfl_xor(r, 1, 32);
        r += __shfl_xor(r, 2, 32);
        r += __shfl_xor(r, 4, 32);
        r += __shfl_xor(r, 8, 32);
        part[i] = r + b3;
      }
      if (nlo == 0) {                    // lane 0 -> rows 0..7, lane 16 -> rows 8..15
#pragma unroll
        for (int i = 0; i < 8; ++i)
          scr[tile * 16 + mbase + i] = part[i];
      }
    }
    __syncthreads();
  }

  // =================== phase 2: decay + softmax + weighted sum ===================
  const float decay = *decayp;
  float val = -INFINITY;
  if (tid < LL) {
    float tw = __expf(decay * (float)(tid - (LL - 1)));
    bool mv = keys_mask[(size_t)b * LL + tid] != 0;
    val = mv ? scr[tid] * tw : -INFINITY;
  }
  __syncthreads();
  red[tid] = val;
  __syncthreads();
  for (int s = 128; s > 0; s >>= 1) {
    if (tid < s) red[tid] = fmaxf(red[tid], red[tid + s]);
    __syncthreads();
  }
  float mx = red[0];
  __syncthreads();
  float e = 0.f;
  if (val > -INFINITY && mx > -INFINITY) e = __expf(val - mx);
  red[tid] = e;
  __syncthreads();
  for (int s = 128; s > 0; s >>= 1) {
    if (tid < s) red[tid] += red[tid + s];
    __syncthreads();
  }
  float sum = red[0];
  __syncthreads();
  float attn = (sum > 0.f) ? (e / sum) : 0.f;
  scr[tid] = attn;
  if (tid < LL)
    out[(size_t)BB * DD + (size_t)b * LL + tid] = attn;
  __syncthreads();

  // weighted_sum[b][d] = sum_l attn[l] * keys[b][l][d]; split L over 2 halves
  const int d = tid & 127, halfsel = tid >> 7;
  float acc = 0.f;
  const int l0 = halfsel * 100;
  for (int l = l0; l < l0 + 100; ++l)
    acc += scr[l] * keys[((size_t)b * LL + l) * DD + d];
  red[tid] = acc;
  __syncthreads();
  if (tid < 128)
    out[(size_t)b * DD + tid] = red[tid] + red[tid + 128];
}

extern "C" void kernel_launch(void* const* d_in, const int* in_sizes, int n_in,
                              void* d_out, int out_size, void* d_ws, size_t ws_size,
                              hipStream_t stream) {
  (void)in_sizes; (void)n_in; (void)out_size; (void)d_ws; (void)ws_size;
  const float*         query = (const float*)d_in[0];
  const float*         keys  = (const float*)d_in[1];
  const unsigned char* mask  = (const unsigned char*)d_in[2];
  const float* W1 = (const float*)d_in[3];
  const float* b1 = (const float*)d_in[4];
  const float* a1 = (const float*)d_in[5];
  const float* W2 = (const float*)d_in[6];
  const float* b2 = (const float*)d_in[7];
  const float* a2 = (const float*)d_in[8];
  const float* W3 = (const float*)d_in[9];
  const float* b3 = (const float*)d_in[10];
  const float* dr = (const float*)d_in[11];
  float* out = (float*)d_out;

  hipFuncSetAttribute((const void*)atda_kernel,
                      hipFuncAttributeMaxDynamicSharedMemorySize, SMEM_BYTES);
  atda_kernel<<<dim3(BB), dim3(256), SMEM_BYTES, stream>>>(
      query, keys, mask, W1, b1, a1, W2, b2, a2, W3, b3, dr, out);
}
